// DifferentiableKernelDensityEstimation_26843545600549
// MI455X (gfx1250) — matrix operand order repaired and checked by reading, not compiled, not run on hardware
//
#include <hip/hip_runtime.h>
#include <hip/hip_bf16.h>

// ---------------------------------------------------------------------------
// KDE: density[n] = mean_m exp( -(||x_n||^2 + ||d_m||^2 - 2 x_n.d_m) * c0 )
//   c0 = 0.5 / sqrt(2*pi)   (h = 1)
// Strategy: f16 WMMA GEMM for the cross term (f32 accumulate), exact f32 for
// the norm terms, native-exp2 epilogue with all constants pre-folded.
// Data staging: double-buffered LDS, async global->LDS (CDNA5) when available.
// ---------------------------------------------------------------------------

typedef float    v8f  __attribute__((ext_vector_type(8)));
typedef _Float16 h16  __attribute__((ext_vector_type(16)));
typedef _Float16 h8   __attribute__((ext_vector_type(8)));
typedef _Float16 h4   __attribute__((ext_vector_type(4)));
typedef unsigned int ui4 __attribute__((ext_vector_type(4)));

// Exact types the async-LDS builtin expects (from the round-2 diagnostic):
// pointer to GCC-vector int4, global (AS1 / __device__) and LDS (AS3) spaces.
typedef int i4 __attribute__((vector_size(16)));
typedef __attribute__((address_space(1))) i4 gi4;
typedef __attribute__((address_space(3))) i4 si4;

#define N_ROWS 8192   // rows of x
#define M_ROWS 8192   // rows of data
#define D_DIM  128    // feature dim (K)

// k = c0 * log2(e) = (0.5/sqrt(2pi)) * 1.4426950408889634
#define KCONST 0.28778707414f
// LDS row stride (halfs): 128 data + 8 pad -> 68 dwords/row, conflict-free
#define LDSSTRIDE 136
#define STAGE_ROWS 64
#define M_SLICE 1024

#if __has_builtin(__builtin_amdgcn_global_load_async_to_lds_b128) && \
    __has_builtin(__builtin_amdgcn_s_wait_asynccnt)
#define USE_ASYNC_LDS 1
#else
#define USE_ASYNC_LDS 0
#endif

static __device__ __forceinline__ float fast_exp2(float x) {
#if __has_builtin(__builtin_amdgcn_exp2f)
  return __builtin_amdgcn_exp2f(x);   // native v_exp_f32
#else
  return exp2f(x);
#endif
}

static __device__ __forceinline__ h16 cat8(h8 lo, h8 hi) {
  return __builtin_shufflevector(lo, hi, 0, 1, 2, 3, 4, 5, 6, 7,
                                         8, 9, 10, 11, 12, 13, 14, 15);
}

// ---------------------------------------------------------------------------
// Prep: f32 -> f16 conversion + per-row sum of squares (one wave32 per row).
// blockIdx.y selects x (0) or data (1). 256 threads = 8 rows per block.
// ---------------------------------------------------------------------------
__global__ __launch_bounds__(256) void kde_prep(
    const float* __restrict__ x, const float* __restrict__ data,
    _Float16* __restrict__ xh, _Float16* __restrict__ dh,
    float* __restrict__ xsq, float* __restrict__ dsq) {
  const int tid  = threadIdx.x;
  const int wave = tid >> 5;
  const int lane = tid & 31;
  const int row  = blockIdx.x * 8 + wave;

  const float* __restrict__ src = blockIdx.y ? data : x;
  _Float16* __restrict__ dsth   = blockIdx.y ? dh : xh;
  float* __restrict__ dstsq     = blockIdx.y ? dsq : xsq;

  const float4 v = *(const float4*)(src + (size_t)row * D_DIM + lane * 4);
  h4 hv;
  hv.x = (_Float16)v.x; hv.y = (_Float16)v.y;
  hv.z = (_Float16)v.z; hv.w = (_Float16)v.w;
  *(h4*)(dsth + (size_t)row * D_DIM + lane * 4) = hv;

  float s = v.x * v.x + v.y * v.y + v.z * v.z + v.w * v.w;
  s += __shfl_xor(s, 16);
  s += __shfl_xor(s, 8);
  s += __shfl_xor(s, 4);
  s += __shfl_xor(s, 2);
  s += __shfl_xor(s, 1);
  if (lane == 0) dstsq[row] = s;
}

// ---------------------------------------------------------------------------
// Cooperative stage copy: 64 data rows x 128 halfs (16 KB) global -> LDS,
// 16-byte transfers, padded LDS row stride.
// ---------------------------------------------------------------------------
static __device__ __forceinline__ void copy_stage(
    _Float16* dst, const _Float16* __restrict__ src, int rowStart, int tid) {
#if USE_ASYNC_LDS
#pragma unroll
  for (int j = 0; j < 4; ++j) {
    const int u = tid + j * 256;      // 0..1023
    const int r = u >> 4;             // row 0..63
    const int c = (u & 15) << 3;      // half col 0..120 step 8
    // CDNA5 async copy: LDS written directly by the memory system (ASYNCcnt).
    __builtin_amdgcn_global_load_async_to_lds_b128(
        (gi4*)(src + (size_t)(rowStart + r) * D_DIM + c),
        (si4*)(dst + r * LDSSTRIDE + c),
        /*offset=*/0, /*cpol=*/0);
  }
#else
#pragma unroll
  for (int j = 0; j < 4; ++j) {
    const int u = tid + j * 256;
    const int r = u >> 4;
    const int c = (u & 15) << 3;
    *(ui4*)(dst + r * LDSSTRIDE + c) =
        *(const ui4*)(src + (size_t)(rowStart + r) * D_DIM + c);
  }
#endif
}

static __device__ __forceinline__ void stage_fence() {
#if USE_ASYNC_LDS
  __builtin_amdgcn_s_wait_asynccnt(0);  // my async LDS writes are done
#endif
}

// ---------------------------------------------------------------------------
// Main: grid (64, 8). Each block: 128 x-rows (8 waves x 16-row A tiles),
// one 1024-row slice of data. A tiles resident in VGPRs for the whole slice.
// Double-buffered LDS staging of data (B) shared by all 8 waves.
// ---------------------------------------------------------------------------
__global__ __launch_bounds__(256) void kde_main(
    const _Float16* __restrict__ xh, const _Float16* __restrict__ dh,
    const float* __restrict__ xsq, const float* __restrict__ dsq,
    float* __restrict__ partial) {
  __shared__ _Float16 smem[2][STAGE_ROWS * LDSSTRIDE];

  const int tid  = threadIdx.x;
  const int w    = tid >> 5;
  const int lane = tid & 31;
  const int l16  = lane & 15;
  const int hi   = lane >> 4;   // 0: lanes 0-15, 1: lanes 16-31
  const int rowBase = blockIdx.x * 128 + w * 16;
  const int mStart  = blockIdx.y * M_SLICE;

  // --- A fragments: 16 x-rows x K=128 as 4 x (16x32 f16) per ISA layout:
  // lane L (group g = L/16): M = L%16; halfs 0-7 -> K = t*32 + 8*g + j,
  // halfs 8-15 -> K = t*32 + 16 + 8*g + (j-8).
  h16 A[4];
  const _Float16* aBase = xh + (size_t)(rowBase + l16) * D_DIM;
#pragma unroll
  for (int t = 0; t < 4; ++t) {
    const h8 lo  = *(const h8*)(aBase + t * 32 + (hi ? 8 : 0));
    const h8 hi8 = *(const h8*)(aBase + t * 32 + (hi ? 24 : 16));
    A[t] = cat8(lo, hi8);
  }

  // C layout: VGPR i <-> x-row (rowBase + i + 8*hi), N = l16.
  float aterm[8];
  const float* xq = xsq + rowBase + (hi ? 8 : 0);
#pragma unroll
  for (int i = 0; i < 8; ++i) aterm[i] = -KCONST * xq[i];

  float acc[8] = {0.f, 0.f, 0.f, 0.f, 0.f, 0.f, 0.f, 0.f};

  copy_stage(&smem[0][0], dh, mStart, tid);

  const int nStages = M_SLICE / STAGE_ROWS;  // 16
  for (int s = 0; s < nStages; ++s) {
    stage_fence();    // wave's async writes to stage s landed in LDS
    __syncthreads();  // all waves' stage-s writes visible; s-1 consumed
    if (s + 1 < nStages)
      copy_stage(&smem[(s + 1) & 1][0], dh, mStart + (s + 1) * STAGE_ROWS, tid);

    const _Float16* buf = &smem[s & 1][0];
#pragma unroll
    for (int g = 0; g < 4; ++g) {
      const int col = mStart + s * STAGE_ROWS + g * 16 + l16;
      const float bterm = -KCONST * dsq[col];
      // B layout (32x16, K x N): lane L: N = L%16; halfs 0-7 -> K = 8*hi + j,
      // halfs 8-15 -> K = 16 + 8*hi + (j-8) (+ t*32 per K tile).
      const _Float16* bRow =
          buf + (g * 16 + l16) * LDSSTRIDE + (hi ? 8 : 0);

      v8f C = {};
#pragma unroll
      for (int t = 0; t < 4; ++t) {
        const h8 blo = *(const h8*)(bRow + t * 32);
        const h8 bhi = *(const h8*)(bRow + t * 32 + 16);
        const h16 B = cat8(blo, bhi);
        C = __builtin_amdgcn_wmma_f32_16x16x32_f16(
            false, A[t], false, B, (short)0, C, false, false);
      }
#pragma unroll
      for (int i = 0; i < 8; ++i)
        acc[i] += fast_exp2(fmaf(C[i], 2.0f * KCONST, aterm[i] + bterm));
    }
  }

  // Deterministic reduce across the 16 lanes holding each row's columns.
#pragma unroll
  for (int i = 0; i < 8; ++i) {
    float v = acc[i];
    v += __shfl_xor(v, 1);
    v += __shfl_xor(v, 2);
    v += __shfl_xor(v, 4);
    v += __shfl_xor(v, 8);
    acc[i] = v;
  }
  if (l16 == 0) {
#pragma unroll
    for (int i = 0; i < 8; ++i)
      partial[(size_t)blockIdx.y * N_ROWS + rowBase + i + 8 * hi] = acc[i];
  }
}

// ---------------------------------------------------------------------------
// Final reduction over the 8 M-slices + scale by 1/M.
// ---------------------------------------------------------------------------
__global__ __launch_bounds__(256) void kde_reduce(
    const float* __restrict__ partial, float* __restrict__ out) {
  const int n = blockIdx.x * 256 + threadIdx.x;
  float s = 0.f;
#pragma unroll
  for (int g = 0; g < 8; ++g) s += partial[(size_t)g * N_ROWS + n];
  out[n] = s * (1.0f / (float)M_ROWS);
}

// ---------------------------------------------------------------------------
extern "C" void kernel_launch(void* const* d_in, const int* in_sizes, int n_in,
                              void* d_out, int out_size, void* d_ws,
                              size_t ws_size, hipStream_t stream) {
  const float* x    = (const float*)d_in[0];
  const float* data = (const float*)d_in[1];
  float* out        = (float*)d_out;

  char* ws = (char*)d_ws;
  const size_t halfBytes = (size_t)N_ROWS * D_DIM * sizeof(_Float16);  // 2 MB
  _Float16* xh   = (_Float16*)ws;
  _Float16* dh   = (_Float16*)(ws + halfBytes);
  float* xsq     = (float*)(ws + 2 * halfBytes);
  float* dsq     = xsq + N_ROWS;
  float* partial = dsq + M_ROWS;  // 8 * 8192 floats

  kde_prep<<<dim3(N_ROWS / 8, 2), 256, 0, stream>>>(x, data, xh, dh, xsq, dsq);
  kde_main<<<dim3(N_ROWS / 128, M_ROWS / M_SLICE), 256, 0, stream>>>(
      xh, dh, xsq, dsq, partial);
  kde_reduce<<<N_ROWS / 256, 256, 0, stream>>>(partial, out);

  (void)in_sizes; (void)n_in; (void)out_size; (void)ws_size;
}
